// MultiheadSelfAttention_74620761800951
// MI455X (gfx1250) — matrix operand order repaired and checked by reading, not compiled, not run on hardware
//
#include <hip/hip_runtime.h>
#include <hip/hip_bf16.h>

// ---------------------------------------------------------------------------
// MI455X (gfx1250) multihead self-attention, bf16 WMMA pipeline.
//   x[2,2048,1024] f32, w_qkv[3072,1024] f32, w_o[1024,1024] f32 -> out f32
// Compute-bound (AI ~1000 flop/B, working set fits 192MB L2);
// all matmuls on v_wmma_f32_16x16x32_bf16 with f32 accumulation.
// ---------------------------------------------------------------------------

typedef __bf16 bf16;
typedef __attribute__((ext_vector_type(16))) __bf16 v16bf;
typedef __attribute__((ext_vector_type(8)))  __bf16 v8bf;
typedef __attribute__((ext_vector_type(4)))  __bf16 v4bf;
typedef __attribute__((ext_vector_type(8)))  float  v8f;
typedef __attribute__((ext_vector_type(4)))  float  v4f;

#define D_MODEL   1024
#define NUM_HEADS 16
#define HEAD_DIM  64
#define BATCH     2
#define SEQ       2048

union V16 { v16bf v; v8bf h[2]; };

static __device__ __forceinline__ v8f wmma_bf16(v16bf a, v16bf b, v8f c) {
    return __builtin_amdgcn_wmma_f32_16x16x32_bf16(false, a, false, b,
                                                   (short)0, c, false, false);
}

// ---------------------------------------------------------------------------
// fp32 -> bf16 cast, 4 elements/thread
// ---------------------------------------------------------------------------
__global__ void cast_f32_bf16_kernel(const float* __restrict__ in,
                                     bf16* __restrict__ out, int n4) {
    int i = blockIdx.x * blockDim.x + threadIdx.x;
    if (i < n4) {
        v4f f = reinterpret_cast<const v4f*>(in)[i];
        v4bf o;
        o[0] = (bf16)f[0]; o[1] = (bf16)f[1];
        o[2] = (bf16)f[2]; o[3] = (bf16)f[3];
        reinterpret_cast<v4bf*>(out)[i] = o;
    }
}

// ---------------------------------------------------------------------------
// C[M,N] = A[M,K] * W[N,K]^T  (bf16 in, OutT out).
// Wave tile 64x64 = 4x4 fragments -> 16 WMMAs per K=32 step vs 16 b128 loads
// (~32.8 flop per L2 byte). Block = 4 waves side-by-side in N -> 64x256 tile.
// ---------------------------------------------------------------------------
template <typename OutT>
__global__ __launch_bounds__(128)
void gemm_bf16_wmma(const bf16* __restrict__ A, const bf16* __restrict__ W,
                    OutT* __restrict__ C, int M, int N, int K) {
    const int lane   = threadIdx.x & 31;
    const int wave   = threadIdx.x >> 5;
    const int half16 = lane >> 4;      // 0: lanes 0-15, 1: lanes 16-31
    const int ln     = lane & 15;

    const int m0 = blockIdx.x * 64;
    const int n0 = blockIdx.y * 256 + wave * 64;

    v8f acc[4][4];
#pragma unroll
    for (int mt = 0; mt < 4; ++mt)
#pragma unroll
        for (int nt = 0; nt < 4; ++nt)
            acc[mt][nt] = (v8f){0.f, 0.f, 0.f, 0.f, 0.f, 0.f, 0.f, 0.f};

    for (int kk = 0; kk < K; kk += 32) {
        // A fragments: 16-bit A 16x32 layout: lane<16 -> row=ln, K {0..7,16..23};
        // lane>=16 -> row=ln, K {8..15,24..31}
        V16 a[4];
#pragma unroll
        for (int mt = 0; mt < 4; ++mt) {
            const bf16* ar = A + (size_t)(m0 + mt * 16 + ln) * K + kk + half16 * 8;
            a[mt].h[0] = *reinterpret_cast<const v8bf*>(ar);
            a[mt].h[1] = *reinterpret_cast<const v8bf*>(ar + 16);
            __builtin_prefetch(ar + 32, 0, 3);  // next K step
        }
        // B fragments: 16-bit B 32x16 layout: lanes0-15 K {0..7,16..23}, lanes16-31 K {8..15,24..31},
        // col = ln.  B(k,n) = W[n, kk+k] -> contiguous along k.
        V16 b[4];
#pragma unroll
        for (int nt = 0; nt < 4; ++nt) {
            const bf16* wr = W + (size_t)(n0 + nt * 16 + ln) * K + kk + half16 * 8;
            b[nt].h[0] = *reinterpret_cast<const v8bf*>(wr);
            b[nt].h[1] = *reinterpret_cast<const v8bf*>(wr + 16);
            __builtin_prefetch(wr + 32, 0, 3);
        }
#pragma unroll
        for (int mt = 0; mt < 4; ++mt)
#pragma unroll
            for (int nt = 0; nt < 4; ++nt)
                acc[mt][nt] = wmma_bf16(a[mt].v, b[nt].v, acc[mt][nt]);
    }

    // C fragment layout: VGPR i -> row m0+mt*16+i (lanes 0-15) / +8 (lanes 16-31)
#pragma unroll
    for (int mt = 0; mt < 4; ++mt)
#pragma unroll
        for (int nt = 0; nt < 4; ++nt)
#pragma unroll
            for (int i = 0; i < 8; ++i) {
                int row = m0 + mt * 16 + i + half16 * 8;
                int col = n0 + nt * 16 + ln;
                C[(size_t)row * N + col] = (OutT)acc[mt][nt][i];
            }
}

// ---------------------------------------------------------------------------
// Causal flash attention.  qkv bf16 [B,S,3*D]; out bf16 [B,S,D].
// Block = 4 waves, 64 query rows of one (b,h); wave owns 16 rows.
// Key blocks of 32; scores via 2x2 WMMA, P*V via 4 WMMA with V^T staged in LDS.
// ---------------------------------------------------------------------------
__global__ __launch_bounds__(128)
void attn_flash_wmma(const bf16* __restrict__ qkv, bf16* __restrict__ out) {
    __shared__ bf16 ldsVT[HEAD_DIM][40];   // V^T tile [hd][key], pad->80B rows
    __shared__ bf16 ldsP[4][16][32];       // per-wave P tile, row-major

    const int lane   = threadIdx.x & 31;
    const int wave   = threadIdx.x >> 5;
    const int half16 = lane >> 4;
    const int ln     = lane & 15;

    const int b    = blockIdx.x / NUM_HEADS;
    const int h    = blockIdx.x % NUM_HEADS;
    const int mg   = blockIdx.y;            // 64-row query group
    const int mrow = mg * 64 + wave * 16;   // this wave's 16 query rows

    const size_t rs = 3 * D_MODEL;          // qkv row stride (elements)
    const bf16* Qb = qkv + (size_t)b * SEQ * rs + h * HEAD_DIM;
    const bf16* Kb = Qb + D_MODEL;
    const bf16* Vb = Qb + 2 * D_MODEL;

    // Q A-fragments for head-dim chunks [0,32) and [32,64)
    V16 aq[2];
#pragma unroll
    for (int c = 0; c < 2; ++c) {
        const bf16* qr = Qb + (size_t)(mrow + ln) * rs + c * 32 + half16 * 8;
        aq[c].h[0] = *reinterpret_cast<const v8bf*>(qr);
        aq[c].h[1] = *reinterpret_cast<const v8bf*>(qr + 16);
    }

    v8f o[4];
#pragma unroll
    for (int nt = 0; nt < 4; ++nt)
        o[nt] = (v8f){0.f, 0.f, 0.f, 0.f, 0.f, 0.f, 0.f, 0.f};
    float mrun[8], lrun[8];
#pragma unroll
    for (int i = 0; i < 8; ++i) { mrun[i] = -3.0e38f; lrun[i] = 0.f; }

    const int nkb = 2 * (mg + 1);           // key blocks needed (causal)
    for (int kb = 0; kb < nkb; ++kb) {
        const int kbase = kb * 32;

        // ---- cooperative transposed V staging (shared by all 4 waves) ----
        __syncthreads();
#pragma unroll
        for (int e = 0; e < 16; ++e) {
            int idx = threadIdx.x + e * 128;        // 0..2047 over 32x64 tile
            int key = idx >> 6;
            int hd  = idx & 63;
            ldsVT[hd][key] = Vb[(size_t)(kbase + key) * rs + hd];
        }
        __syncthreads();

        // ---- scores: S = Q(16x64) . K^T, two 16-key tiles ----
        v8f s[2];
#pragma unroll
        for (int t = 0; t < 2; ++t) {
            const bf16* kr = Kb + (size_t)(kbase + t * 16 + ln) * rs + half16 * 8;
            V16 bk0, bk1;                 // head-dim chunks 0 and 1
            bk0.h[0] = *reinterpret_cast<const v8bf*>(kr);
            bk0.h[1] = *reinterpret_cast<const v8bf*>(kr + 16);
            bk1.h[0] = *reinterpret_cast<const v8bf*>(kr + 32);
            bk1.h[1] = *reinterpret_cast<const v8bf*>(kr + 48);
            v8f z = (v8f){0.f, 0.f, 0.f, 0.f, 0.f, 0.f, 0.f, 0.f};
            z = wmma_bf16(aq[0].v, bk0.v, z);
            z = wmma_bf16(aq[1].v, bk1.v, z);
            s[t] = z;
        }

        // ---- scale + causal mask + online softmax (row reductions width=16) ----
        float alpha[8];
#pragma unroll
        for (int i = 0; i < 8; ++i) {
            const int qrow = mrow + i + half16 * 8;
#pragma unroll
            for (int t = 0; t < 2; ++t) {
                int key = kbase + t * 16 + ln;
                float v = s[t][i] * 0.125f;          // 1/sqrt(64)
                s[t][i] = (key <= qrow) ? v : -3.0e38f;
            }
            float mx = fmaxf(s[0][i], s[1][i]);
#pragma unroll
            for (int d = 1; d < 16; d <<= 1) mx = fmaxf(mx, __shfl_xor(mx, d, 16));
            float mnew = fmaxf(mrun[i], mx);
            float a_   = __expf(mrun[i] - mnew);
            mrun[i]    = mnew;
            float p0 = __expf(s[0][i] - mnew);
            float p1 = __expf(s[1][i] - mnew);
            s[0][i] = p0; s[1][i] = p1;
            float rsum = p0 + p1;
#pragma unroll
            for (int d = 1; d < 16; d <<= 1) rsum += __shfl_xor(rsum, d, 16);
            lrun[i] = lrun[i] * a_ + rsum;
            alpha[i] = a_;
        }
#pragma unroll
        for (int nt = 0; nt < 4; ++nt)
#pragma unroll
            for (int i = 0; i < 8; ++i) o[nt][i] *= alpha[i];

        // ---- remap P (C layout) -> A layout via per-wave LDS ----
#pragma unroll
        for (int i = 0; i < 8; ++i) {
            int r = i + half16 * 8;
            ldsP[wave][r][ln]      = (bf16)s[0][i];
            ldsP[wave][r][16 + ln] = (bf16)s[1][i];
        }
        __builtin_amdgcn_wave_barrier();
        asm volatile("s_wait_dscnt 0" ::: "memory");
        V16 ap;
        {
            const bf16* pr = &ldsP[wave][ln][half16 * 8];
            ap.h[0] = *reinterpret_cast<const v8bf*>(pr);
            ap.h[1] = *reinterpret_cast<const v8bf*>(pr + 16);
        }

        // ---- O += P(16x32) . V(32x64) : 4 accumulating WMMAs ----
#pragma unroll
        for (int nt = 0; nt < 4; ++nt) {
            const bf16* vr = &ldsVT[nt * 16 + ln][half16 * 8];
            V16 bv;
            bv.h[0] = *reinterpret_cast<const v8bf*>(vr);
            bv.h[1] = *reinterpret_cast<const v8bf*>(vr + 16);
            o[nt] = wmma_bf16(ap.v, bv.v, o[nt]);
        }
    }

    // ---- epilogue: normalize, write [b, row, h*64 + hd] (head-merged) ----
#pragma unroll
    for (int nt = 0; nt < 4; ++nt)
#pragma unroll
        for (int i = 0; i < 8; ++i) {
            int row = mrow + i + half16 * 8;
            int col = h * HEAD_DIM + nt * 16 + ln;
            out[(size_t)(b * SEQ + row) * D_MODEL + col] = (bf16)(o[nt][i] / lrun[i]);
        }
}

// ---------------------------------------------------------------------------
// Host-side launch
// ---------------------------------------------------------------------------
extern "C" void kernel_launch(void* const* d_in, const int* in_sizes, int n_in,
                              void* d_out, int out_size, void* d_ws, size_t ws_size,
                              hipStream_t stream) {
    const float* x     = (const float*)d_in[0];   // [2,2048,1024]
    const float* w_qkv = (const float*)d_in[1];   // [3072,1024]
    const float* w_o   = (const float*)d_in[2];   // [1024,1024]
    float*       out   = (float*)d_out;           // [2,2048,1024]

    const int M  = BATCH * SEQ;        // 4096
    const int K  = D_MODEL;            // 1024
    const int N3 = 3 * D_MODEL;        // 3072

    // Workspace layout (bf16 buffers), ~48 MB total
    char* ws = (char*)d_ws;
    bf16* xb    = (bf16*)(ws);                         //  8 MB: x bf16
    bf16* wqkvb = (bf16*)(ws + 8388608);               //  6 MB: w_qkv bf16
    bf16* wob   = (bf16*)(ws + 14680064);              //  2 MB: w_o bf16
    bf16* qkvb  = (bf16*)(ws + 16777216);              // 24 MB: qkv bf16
    bf16* attnb = (bf16*)(ws + 41943040);              //  8 MB: attn out bf16

    // 1. fp32 -> bf16 casts
    {
        int n4 = (M * K) / 4;          // 1048576
        cast_f32_bf16_kernel<<<n4 / 256, 256, 0, stream>>>(x, xb, n4);
        n4 = (N3 * K) / 4;             // 786432
        cast_f32_bf16_kernel<<<n4 / 256, 256, 0, stream>>>(w_qkv, wqkvb, n4);
        n4 = (K * K) / 4;              // 262144
        cast_f32_bf16_kernel<<<n4 / 256, 256, 0, stream>>>(w_o, wob, n4);
    }

    // 2. QKV projection: qkv = x . w_qkv^T  (4096 x 3072 x 1024)
    {
        dim3 grid(M / 64, N3 / 256);   // (64, 12)
        gemm_bf16_wmma<bf16><<<grid, 128, 0, stream>>>(xb, wqkvb, qkvb, M, N3, K);
    }

    // 3. Causal flash attention per (b,h), 64 query rows per block
    {
        dim3 grid(BATCH * NUM_HEADS, SEQ / 64);   // (32, 32)
        attn_flash_wmma<<<grid, 128, 0, stream>>>(qkvb, attnb);
    }

    // 4. Output projection: out = attn . w_o^T  (4096 x 1024 x 1024), f32 out
    {
        dim3 grid(M / 64, K / 256);    // (64, 4)
        gemm_bf16_wmma<float><<<grid, 128, 0, stream>>>(attnb, wob, out, M, K, K);
    }
}
